// ParticleNetForFineTune_76819785056878
// MI455X (gfx1250) — compile-verified, matrix operand order checked by reading, not run on hardware
//
#include <hip/hip_runtime.h>
#include <hip/hip_bf16.h>

// ---------------------------------------------------------------------------
// ParticleNet forward for MI455X (gfx1250, wave32).
// All heavy GEMMs use V_WMMA_F32_16X16X32_BF16 (f32 accumulate).
// ---------------------------------------------------------------------------

typedef __attribute__((ext_vector_type(16))) __bf16 bf16x16;
typedef __attribute__((ext_vector_type(8)))  float  f32x8;

namespace {
constexpr int  BATCH = 32;
constexpr int  NPTS  = 256;
constexpr int  KNBR  = 16;
constexpr int  PTS   = BATCH * NPTS;        // 8192 points
constexpr long EROWS = (long)PTS * KNBR;    // 131072 edge rows
}

union FragU { uint4 u4[2]; bf16x16 v; };
union Pack4 { __hip_bfloat16 h[4]; uint2 u; };

// Load a 16x32 bf16 fragment (A-matrix layout per CDNA5 ISA 7.12.2):
// p must already point at row*ld + k0 + half*8; chunks at p and p+16.
// lanes 0-15 row M=lane: K {k0..k0+7} in v0-3, {k0+16..k0+23} in v4-7;
// lanes 16-31: K {k0+8..k0+15} and {k0+24..k0+31}.  Two 16B loads per lane.
__device__ __forceinline__ bf16x16 load_frag(const __hip_bfloat16* p) {
  FragU f;
  f.u4[0] = *(const uint4*)(p);
  f.u4[1] = *(const uint4*)(p + 16);
  return f.v;
}

// ---------------------------------------------------------------------------
// WMMA GEMM: D[m,o] = act( scale[o]*sum_k A[m,k]*W[o,k] + bias[o] , ... )
// A: M x K bf16 row-major.  W: O x K bf16 row-major (== B^T layout WMMA wants).
// One wave computes a 32(M) x 64(O) tile: 2 A frags x 4 W frags = 8 WMMAs per
// K-step, all 6 fragment loads issued BEFORE the WMMA block so the scheduler
// can clause the loads and overlap them with matrix math.
// mode 0: y=relu(bn(y))            -> out_bf (M x O bf16)
// mode 1: v=relu(bn(y)+agg)*mask   -> out_f32 (M x O) and fused slice out_bf
// mode 2: y=relu(bn(y))*mask       -> out_f32 (M x O)
// ---------------------------------------------------------------------------
__global__ __launch_bounds__(256) void wmma_gemm_kernel(
    const __hip_bfloat16* __restrict__ A,
    const __hip_bfloat16* __restrict__ W,
    const float* __restrict__ scale, const float* __restrict__ bias,
    int M, int K, int O, int mode,
    __hip_bfloat16* __restrict__ out_bf,
    float* __restrict__ out_f32,
    const float* __restrict__ agg,
    const float* __restrict__ maskv,
    int ldf, int coff) {
  const int lane = threadIdx.x & 31;
  const int wave = (int)((blockIdx.x * blockDim.x + threadIdx.x) >> 5);
  const int l = lane & 15, half = lane >> 4;
  const int ngrp = O >> 6;                 // O is always a multiple of 64
  const int mt = wave / ngrp;
  const int ng = wave - mt * ngrp;
  const int m0 = mt * 32;
  if (m0 >= M) return;                     // wave-uniform: EXEC stays all-ones
  const int n0 = ng * 64;

  // Per-lane base pointers (include the half-dependent +8 element offset).
  const __hip_bfloat16* ap0 = A + (size_t)(m0 + l) * K + half * 8;
  const __hip_bfloat16* ap1 = A + (size_t)(m0 + 16 + l) * K + half * 8;
  const __hip_bfloat16* wp[4];
#pragma unroll
  for (int t = 0; t < 4; ++t) wp[t] = W + (size_t)(n0 + t * 16 + l) * K + half * 8;

  f32x8 acc[2][4] = {};
  for (int k0 = 0; k0 < K; k0 += 32) {
    // ---- load phase: 12 x b128 (clause-able) ----
    bf16x16 a0 = load_frag(ap0 + k0);
    bf16x16 a1 = load_frag(ap1 + k0);
    bf16x16 wf[4];
#pragma unroll
    for (int t = 0; t < 4; ++t) wf[t] = load_frag(wp[t] + k0);
    if (k0 + 32 < K) {                     // prefetch streaming A operand
      __builtin_prefetch(ap0 + k0 + 32, 0, 3);
      __builtin_prefetch(ap1 + k0 + 32, 0, 3);
    }
    // ---- compute phase: 8 back-to-back WMMAs ----
#pragma unroll
    for (int t = 0; t < 4; ++t) {
      acc[0][t] = __builtin_amdgcn_wmma_f32_16x16x32_bf16(
          false, a0, false, wf[t], (short)0, acc[0][t], false, false);
      acc[1][t] = __builtin_amdgcn_wmma_f32_16x16x32_bf16(
          false, a1, false, wf[t], (short)0, acc[1][t], false, false);
    }
  }

  // C/D layout: lane -> N (l), half selects M+8; VGPR r -> M = base + r + half*8.
#pragma unroll
  for (int s = 0; s < 2; ++s) {
#pragma unroll
    for (int t = 0; t < 4; ++t) {
      const int o = n0 + t * 16 + l;
      const float sc = scale[o], bb = bias[o];
#pragma unroll
      for (int r = 0; r < 8; ++r) {
        const int m = m0 + s * 16 + r + half * 8;
        float y = acc[s][t][r] * sc + bb;
        if (mode == 0) {
          y = fmaxf(y, 0.f);
          out_bf[(size_t)m * O + o] = __float2bfloat16(y);
        } else if (mode == 1) {
          float v = y + agg[(size_t)m * O + o];
          v = fmaxf(v, 0.f) * maskv[m];
          out_f32[(size_t)m * O + o] = v;
          out_bf[(size_t)m * ldf + coff + o] = __float2bfloat16(v);
        } else {
          y = fmaxf(y, 0.f) * maskv[m];
          out_f32[(size_t)m * O + o] = y;
        }
      }
    }
  }
}

// --------------------------- support kernels -------------------------------

// Fold BN: y = x*s + t with s = g*rsqrt(v+eps), t = b - m*s
__global__ void bn_fold_kernel(const float* __restrict__ gg, const float* __restrict__ bb,
                               const float* __restrict__ mm, const float* __restrict__ vv,
                               float* __restrict__ s, float* __restrict__ t, int n) {
  int i = blockIdx.x * blockDim.x + threadIdx.x;
  if (i >= n) return;
  float sc = gg[i] * rsqrtf(vv[i] + 1e-5f);
  s[i] = sc;
  t[i] = bb[i] - mm[i] * sc;
}

// f32 -> bf16, 4 elements per thread (n4 = n/4; all call sites are %4==0).
__global__ void f2bf_kernel(const float* __restrict__ src,
                            __hip_bfloat16* __restrict__ dst, long n4) {
  long i = (long)blockIdx.x * blockDim.x + threadIdx.x;
  if (i >= n4) return;
  float4 v = ((const float4*)src)[i];
  Pack4 o;
  o.h[0] = __float2bfloat16(v.x);
  o.h[1] = __float2bfloat16(v.y);
  o.h[2] = __float2bfloat16(v.z);
  o.h[3] = __float2bfloat16(v.w);
  ((uint2*)dst)[i] = o.u;
}

// Masked points (B,N,2) and BN(features)*mask as (B,N,32) row-major.
__global__ void prep_kernel(const float* __restrict__ points, const float* __restrict__ feats,
                            const float* __restrict__ mask,
                            const float* __restrict__ s_fts, const float* __restrict__ t_fts,
                            float* __restrict__ pts0, float* __restrict__ fts0) {
  int g = blockIdx.x * blockDim.x + threadIdx.x;
  if (g >= PTS) return;
  const int b = g >> 8, n = g & 255;
  const float mk = mask[g];
  pts0[g * 2 + 0] = points[(b * 2 + 0) * NPTS + n] * mk;
  pts0[g * 2 + 1] = points[(b * 2 + 1) * NPTS + n] * mk;
  for (int c = 0; c < 32; ++c) {
    float x = feats[((size_t)b * 32 + c) * NPTS + n] * mk;
    fts0[(size_t)g * 32 + c] = (x * s_fts[c] + t_fts[c]) * mk;
  }
}

__global__ void counts_kernel(const float* __restrict__ mask, float* __restrict__ counts) {
  int b = threadIdx.x;
  if (b >= BATCH) return;
  float s = 0.f;
  for (int n = 0; n < NPTS; ++n) s += mask[b * NPTS + n];
  counts[b] = fmaxf(s, 1.0f);
}

// Top-16 nearest neighbors per point (self excluded like the reference: keep
// the 17 smallest (d, idx) with stable tie order, drop rank 0).  Insertion via
// fully-unrolled compare-exchange so state stays in registers.
__global__ void knn_kernel(const float* __restrict__ pts, int D,
                           const float* __restrict__ mask,
                           int* __restrict__ nidx) {
  int g = blockIdx.x * blockDim.x + threadIdx.x;
  if (g >= PTS) return;
  const int b = g >> 8;
  const float* pn = pts + (size_t)g * D;
  const float sn = (mask[g] == 0.f) ? 1e9f : 0.f;
  float bd[KNBR + 1];
  int   bi[KNBR + 1];
#pragma unroll
  for (int j = 0; j <= KNBR; ++j) { bd[j] = 3.0e38f; bi[j] = 0; }
  const float* pb = pts + (size_t)b * NPTS * D;
  const float* mb = mask + b * NPTS;
  for (int m = 0; m < NPTS; ++m) {
    const float* pm = pb + (size_t)m * D;
    const float sm = (mb[m] == 0.f) ? 1e9f : 0.f;
    const float ds = sn - sm;
    float d = 0.f;
    for (int c = 0; c < D; ++c) { float t = pn[c] - pm[c] + ds; d += t * t; }
    float cd = d; int ci = m;
#pragma unroll
    for (int j = 0; j <= KNBR; ++j) {
      if (cd < bd[j]) {
        float td = bd[j]; int ti = bi[j];
        bd[j] = cd; bi[j] = ci; cd = td; ci = ti;
      }
    }
  }
#pragma unroll
  for (int j = 0; j < KNBR; ++j) nidx[(size_t)g * KNBR + j] = bi[j + 1];
}

// Edge features [center, nbr-center] as bf16, EROWS x 2C row-major.
// 4 channels per thread: float4 loads, packed 4xbf16 (8B) store.
__global__ void gather_edge_kernel(const float* __restrict__ fts, const int* __restrict__ nidx,
                                   __hip_bfloat16* __restrict__ edge, int C) {
  const int grp = C >> 1;                    // (2C)/4 groups per row
  long i = (long)blockIdx.x * blockDim.x + threadIdx.x;
  const long total = EROWS * (long)grp;
  if (i >= total) return;
  const long row = i / grp;
  const int c4 = (int)(i - row * grp) * 4;
  const int j = (int)(row & (KNBR - 1));
  const long pn = row >> 4;                  // b*N + n
  float4 v;
  if (c4 < C) {
    v = *(const float4*)(fts + pn * C + c4);
  } else {
    const int c = c4 - C;
    const int nb = nidx[pn * KNBR + j];
    const long pm = ((pn >> 8) << 8) + nb;   // b*N + neighbor
    float4 a = *(const float4*)(fts + pm * C + c);
    float4 b = *(const float4*)(fts + pn * C + c);
    v.x = a.x - b.x; v.y = a.y - b.y; v.z = a.z - b.z; v.w = a.w - b.w;
  }
  Pack4 o;
  o.h[0] = __float2bfloat16(v.x);
  o.h[1] = __float2bfloat16(v.y);
  o.h[2] = __float2bfloat16(v.z);
  o.h[3] = __float2bfloat16(v.w);
  *(uint2*)(edge + row * (2L * C) + c4) = o.u;
}

// Mean over k neighbors: (EROWS x O bf16) -> (PTS x O f32), 4 outputs/thread.
__global__ void mean_k_kernel(const __hip_bfloat16* __restrict__ conv,
                              float* __restrict__ agg, int O) {
  const int og = O >> 2;
  long i = (long)blockIdx.x * blockDim.x + threadIdx.x;
  if (i >= (long)PTS * og) return;
  const long pn = i / og;
  const int o4 = (int)(i - pn * og) * 4;
  float s0 = 0.f, s1 = 0.f, s2 = 0.f, s3 = 0.f;
#pragma unroll
  for (int j = 0; j < KNBR; ++j) {
    Pack4 d;
    d.u = *(const uint2*)(conv + (pn * KNBR + j) * (long)O + o4);
    s0 += __bfloat162float(d.h[0]);
    s1 += __bfloat162float(d.h[1]);
    s2 += __bfloat162float(d.h[2]);
    s3 += __bfloat162float(d.h[3]);
  }
  float4 r;
  r.x = s0 * (1.0f / KNBR); r.y = s1 * (1.0f / KNBR);
  r.z = s2 * (1.0f / KNBR); r.w = s3 * (1.0f / KNBR);
  *(float4*)(agg + pn * O + o4) = r;
}

// Count-normalized pooling: x[b,o] = sum_n fus[b,n,o] / counts[b]
__global__ void pool_kernel(const float* __restrict__ fus, const float* __restrict__ counts,
                            float* __restrict__ x) {
  int i = blockIdx.x * blockDim.x + threadIdx.x;
  if (i >= BATCH * 384) return;
  int b = i / 384, o = i - b * 384;
  float s = 0.f;
  for (int n = 0; n < NPTS; ++n) s += fus[((size_t)b * NPTS + n) * 384 + o];
  x[i] = s / counts[b];
}

__global__ void fc1_kernel(const float* __restrict__ x, const float* __restrict__ w,
                           const float* __restrict__ bias, float* __restrict__ h) {
  int i = blockIdx.x * blockDim.x + threadIdx.x;
  if (i >= BATCH * 256) return;
  int b = i / 256, o = i - b * 256;
  float s = bias[o];
  for (int c = 0; c < 384; ++c) s += x[b * 384 + c] * w[o * 384 + c];
  h[i] = fmaxf(s, 0.f);
}

__global__ void fc2_kernel(const float* __restrict__ h, const float* __restrict__ w,
                           const float* __restrict__ bias, float* __restrict__ out) {
  int i = blockIdx.x * blockDim.x + threadIdx.x;
  if (i >= BATCH * 10) return;
  int b = i / 10, o = i - b * 10;
  float s = bias[o];
  for (int c = 0; c < 256; ++c) s += h[b * 256 + c] * w[o * 256 + c];
  out[i] = s;
}

// ---------------------------------------------------------------------------
// Host orchestration.
// Input flattening (setup_inputs dict order; params as jax pytree with dict
// keys sorted alphabetically):
//   0 points  1 features  2 mask
//   3..6   bn_fts        b,g,m,v (32)
//   convs[l] (l = 0..2, keys: bns, sc_bn, sc_w, ws):
//     base+0..11  bns[0..2] each b,g,m,v
//     base+12..15 sc_bn b,g,m,v
//     base+16     sc_w   (chan x cin)
//     base+17..19 ws[0..2]
//     bases: 7, 27, 47
//   67 fc1_b  68 fc1_w  69 fc2_b  70 fc2_w
//   71..74 fusion_bn b,g,m,v  75 fusion_w (384x448)
// ---------------------------------------------------------------------------
extern "C" void kernel_launch(void* const* d_in, const int* in_sizes, int n_in,
                              void* d_out, int out_size, void* d_ws, size_t ws_size,
                              hipStream_t stream) {
  (void)in_sizes; (void)n_in; (void)out_size; (void)ws_size;
  const float* points = (const float*)d_in[0];
  const float* feats  = (const float*)d_in[1];
  const float* mask   = (const float*)d_in[2];
  auto F = [&](int i) { return (const float*)d_in[i]; };

  // bump allocator over d_ws (~190 MB total; recomputed identically each call)
  char* wp = (char*)d_ws;
  auto alloc = [&](size_t bytes) -> void* {
    void* r = (void*)wp;
    wp += (bytes + 255) & ~(size_t)255;
    return r;
  };
  auto gridFor = [](long n) { return dim3((unsigned)((n + 255) / 256)); };

  const int chan[3] = {64, 128, 256};
  const int cin[3]  = {32, 64, 128};
  const int bnsb[3] = {7, 27, 47};
  const int foff[3] = {0, 64, 192};        // fused concat column offsets (448 total)

  // folded BN params + bf16 weights
  float *s_fts = (float*)alloc(32 * 4), *t_fts = (float*)alloc(32 * 4);
  float *s_conv[3][3], *t_conv[3][3], *s_sc[3], *t_sc[3];
  __hip_bfloat16 *wbf[3][3], *scbf[3];
  for (int l = 0; l < 3; ++l) {
    for (int t = 0; t < 3; ++t) {
      s_conv[l][t] = (float*)alloc(chan[l] * 4);
      t_conv[l][t] = (float*)alloc(chan[l] * 4);
      int Kt = (t == 0) ? 2 * cin[l] : chan[l];
      wbf[l][t] = (__hip_bfloat16*)alloc((size_t)chan[l] * Kt * 2);
    }
    s_sc[l] = (float*)alloc(chan[l] * 4);
    t_sc[l] = (float*)alloc(chan[l] * 4);
    scbf[l] = (__hip_bfloat16*)alloc((size_t)chan[l] * cin[l] * 2);
  }
  float* s_fus = (float*)alloc(384 * 4);
  float* t_fus = (float*)alloc(384 * 4);
  __hip_bfloat16* fusw = (__hip_bfloat16*)alloc((size_t)384 * 448 * 2);

  // activations / staging
  float* pts0   = (float*)alloc((size_t)PTS * 2 * 4);
  float* ftsA   = (float*)alloc((size_t)PTS * 256 * 4);
  float* ftsB   = (float*)alloc((size_t)PTS * 256 * 4);
  __hip_bfloat16* fts_bf = (__hip_bfloat16*)alloc((size_t)PTS * 256 * 2);
  int*   knn    = (int*)alloc((size_t)PTS * KNBR * 4);
  __hip_bfloat16* bufA = (__hip_bfloat16*)alloc((size_t)EROWS * 256 * 2);  // 67 MB
  __hip_bfloat16* bufB = (__hip_bfloat16*)alloc((size_t)EROWS * 256 * 2);  // 67 MB
  float* agg    = (float*)alloc((size_t)PTS * 256 * 4);
  __hip_bfloat16* fused = (__hip_bfloat16*)alloc((size_t)PTS * 448 * 2);
  float* fusout = (float*)alloc((size_t)PTS * 384 * 4);
  float* xpool  = (float*)alloc((size_t)BATCH * 384 * 4);
  float* hbuf   = (float*)alloc((size_t)BATCH * 256 * 4);
  float* counts = (float*)alloc(BATCH * 4);

  auto launch_gemm = [&](const __hip_bfloat16* A, const __hip_bfloat16* W,
                         const float* s, const float* t, int M, int K, int O,
                         int mode, __hip_bfloat16* obf, float* of32,
                         const float* aggp, const float* mk, int ldf, int coff) {
    long waves = (long)(M / 32) * (O / 64);  // 32x64 tile per wave
    unsigned blocks = (unsigned)((waves + 7) / 8);   // 8 waves / 256-thread block
    wmma_gemm_kernel<<<blocks, 256, 0, stream>>>(A, W, s, t, M, K, O, mode,
                                                 obf, of32, aggp, mk, ldf, coff);
  };

  // ---- fold BN + convert weights to bf16 (all independent) ----
  bn_fold_kernel<<<gridFor(32), 256, 0, stream>>>(F(4), F(3), F(5), F(6), s_fts, t_fts, 32);
  for (int l = 0; l < 3; ++l) {
    const int O = chan[l], C = cin[l];
    for (int t = 0; t < 3; ++t) {
      int base = bnsb[l] + t * 4;   // b,g,m,v
      bn_fold_kernel<<<gridFor(O), 256, 0, stream>>>(F(base + 1), F(base + 0),
                                                     F(base + 2), F(base + 3),
                                                     s_conv[l][t], t_conv[l][t], O);
      int Kt = (t == 0) ? 2 * C : O;
      f2bf_kernel<<<gridFor((long)O * Kt / 4), 256, 0, stream>>>(F(bnsb[l] + 17 + t),
                                                                 wbf[l][t], (long)O * Kt / 4);
    }
    int scb = bnsb[l] + 12;
    bn_fold_kernel<<<gridFor(O), 256, 0, stream>>>(F(scb + 1), F(scb + 0),
                                                   F(scb + 2), F(scb + 3),
                                                   s_sc[l], t_sc[l], O);
    f2bf_kernel<<<gridFor((long)O * C / 4), 256, 0, stream>>>(F(bnsb[l] + 16), scbf[l],
                                                              (long)O * C / 4);
  }
  bn_fold_kernel<<<gridFor(384), 256, 0, stream>>>(F(72), F(71), F(73), F(74),
                                                   s_fus, t_fus, 384);
  f2bf_kernel<<<gridFor((long)384 * 448 / 4), 256, 0, stream>>>(F(75), fusw,
                                                                (long)384 * 448 / 4);

  // ---- preprocessing ----
  prep_kernel<<<gridFor(PTS), 256, 0, stream>>>(points, feats, mask, s_fts, t_fts,
                                                pts0, ftsA);
  counts_kernel<<<1, 32, 0, stream>>>(mask, counts);

  // ---- EdgeConv blocks ----
  float* fts_cur = ftsA;
  float* fts_nxt = ftsB;
  int C = 32;
  for (int l = 0; l < 3; ++l) {
    const int O = chan[l];
    const float* ptsrc = (l == 0) ? pts0 : fts_cur;   // coord_shift applied in-kernel
    const int D = (l == 0) ? 2 : C;
    knn_kernel<<<gridFor(PTS), 256, 0, stream>>>(ptsrc, D, mask, knn);
    f2bf_kernel<<<gridFor((long)PTS * C / 4), 256, 0, stream>>>(fts_cur, fts_bf,
                                                                (long)PTS * C / 4);
    gather_edge_kernel<<<gridFor(EROWS * (long)(C >> 1)), 256, 0, stream>>>(fts_cur, knn,
                                                                            bufA, C);
    // 3 pointwise convs on 131072 edge rows (BN+ReLU fused)
    launch_gemm(bufA, wbf[l][0], s_conv[l][0], t_conv[l][0], (int)EROWS, 2 * C, O,
                0, bufB, nullptr, nullptr, nullptr, 0, 0);
    launch_gemm(bufB, wbf[l][1], s_conv[l][1], t_conv[l][1], (int)EROWS, O, O,
                0, bufA, nullptr, nullptr, nullptr, 0, 0);
    launch_gemm(bufA, wbf[l][2], s_conv[l][2], t_conv[l][2], (int)EROWS, O, O,
                0, bufB, nullptr, nullptr, nullptr, 0, 0);
    mean_k_kernel<<<gridFor((long)PTS * (O >> 2)), 256, 0, stream>>>(bufB, agg, O);
    // shortcut GEMM; epilogue: relu(bn(sc)+agg)*mask -> next fts (f32) + fused slice
    launch_gemm(fts_bf, scbf[l], s_sc[l], t_sc[l], PTS, C, O,
                1, fused, fts_nxt, agg, mask, 448, foff[l]);
    float* tmp = fts_cur; fts_cur = fts_nxt; fts_nxt = tmp;
    C = O;
  }

  // ---- fusion conv (448 -> 384), relu(bn)*mask ----
  launch_gemm(fused, fusw, s_fus, t_fus, PTS, 448, 384,
              2, nullptr, fusout, nullptr, mask, 0, 0);

  // ---- pooling + FC head ----
  pool_kernel<<<gridFor(BATCH * 384), 256, 0, stream>>>(fusout, counts, xpool);
  fc1_kernel<<<gridFor(BATCH * 256), 256, 0, stream>>>(xpool, F(68), F(67), hbuf);
  fc2_kernel<<<gridFor(BATCH * 10), 256, 0, stream>>>(hbuf, F(70), F(69), (float*)d_out);
}
